// GIN_1769526526271
// MI455X (gfx1250) — matrix operand order repaired and checked by reading, not compiled, not run on hardware
//
#include <hip/hip_runtime.h>
#include <hip/hip_bf16.h>

// ---------------------------------------------------------------------------
// GIN forward for gfx1250 (MI455X): bf16 WMMA GEMMs, f32 atomic scatter-add.
// ---------------------------------------------------------------------------

typedef __attribute__((ext_vector_type(16))) __bf16 v16bf;
typedef __attribute__((ext_vector_type(8)))  __bf16 v8bf;
typedef __attribute__((ext_vector_type(8)))  float  v8f;

#define DH 128
#define BN_EPS 1e-5f

__device__ __forceinline__ unsigned short f2bf(float f) {
    unsigned int u = __float_as_uint(f);
    unsigned int r = 0x7FFFu + ((u >> 16) & 1u);   // round-to-nearest-even
    return (unsigned short)((u + r) >> 16);
}

// -------------------------------- zero -------------------------------------
__global__ void k_zero(float* __restrict__ p, long n) {
    long i = (long)blockIdx.x * blockDim.x + threadIdx.x;
    if (i < n) p[i] = 0.0f;
}

// ------------------------- edge scatter (agg += h[col]) --------------------
// 32 lanes per edge, float4 per lane: coalesced read, 4 f32 atomics.
__global__ void k_scatter(const float* __restrict__ h, const int* __restrict__ row,
                          const int* __restrict__ col, float* __restrict__ agg, int E) {
    long tid = (long)blockIdx.x * blockDim.x + threadIdx.x;
    int e = (int)(tid >> 5);
    if (e >= E) return;
    int j = (int)(tid & 31) << 2;
    int s = col[e], d = row[e];
    const float4 v = *(const float4*)(h + (long)s * DH + j);
    float* dst = agg + (long)d * DH + j;
    atomicAdd(dst + 0, v.x);
    atomicAdd(dst + 1, v.y);
    atomicAdd(dst + 2, v.z);
    atomicAdd(dst + 3, v.w);
}

// ------------------- z0 = (1+eps)*h + agg  -> bf16 -------------------------
__global__ void k_fuse_eps(const float* __restrict__ h, const float* __restrict__ agg,
                           const float* __restrict__ eps, unsigned short* __restrict__ a,
                           long n) {
    long i = (long)blockIdx.x * blockDim.x + threadIdx.x;
    if (i >= n) return;
    float v = (1.0f + eps[0]) * h[i] + agg[i];
    a[i] = f2bf(v);
}

// --------------- W (f32, [k][c]) -> WT (bf16, [c][k]) ----------------------
__global__ void k_cvt_wt(const float* __restrict__ W, unsigned short* __restrict__ WT) {
    int i = blockIdx.x * blockDim.x + threadIdx.x;   // 16384
    int k = i >> 7, c = i & 127;
    WT[c * DH + k] = f2bf(W[i]);
}

// ------------------------- WMMA GEMM: Z = A @ W + b ------------------------
// A: [N][128] bf16 row-major.  WT: [128][128] bf16 = W transposed ([col][k]).
// Block = 128 threads = 4 waves; wave w owns rows [blk*128 + w*32, +32)
// as TWO 16-row tiles sharing each B fragment (2 independent WMMA chains).
// Per wave: 8 N-tiles x 4 K-chunks x 2 row-tiles = 64 v_wmma_f32_16x16x32_bf16.
__global__ void __launch_bounds__(128)
k_gemm128(const unsigned short* __restrict__ A, const unsigned short* __restrict__ WT,
          const float* __restrict__ bias, float* __restrict__ Z, int N) {
    __shared__ unsigned short sB[DH * DH];           // 32 KB: whole weight matrix

    // cooperative coalesced LDS fill (uint4 = 16 B chunks)
    {
        const uint4* src = (const uint4*)WT;
        uint4* dst = (uint4*)sB;
        for (int i = threadIdx.x; i < (DH * DH) / 8; i += 128) dst[i] = src[i];
    }
    __syncthreads();

    const int lane = threadIdx.x & 31;
    const int wid  = threadIdx.x >> 5;
    const int half = lane >> 4;                      // 0 | 1
    const int l16  = lane & 15;
    const int r0   = blockIdx.x * 128 + wid * 32;    // tile 0 rows
    const int r1   = r0 + 16;                        // tile 1 rows
    const bool fullblk = (blockIdx.x * 128 + 128) <= N;   // uniform fast path

    // --- A fragments (16x32 bf16, contiguous k = 16*half + s, same as B), 4 K-chunks x 2 tiles, in regs.
    int ar0 = r0 + l16; if (ar0 >= N) ar0 = N - 1;   // clamp (stores guarded)
    int ar1 = r1 + l16; if (ar1 >= N) ar1 = N - 1;
    const unsigned short* ap0 = A + ar0 * DH;
    const unsigned short* ap1 = A + ar1 * DH;
    v16bf afr0[4], afr1[4];
#pragma unroll
    for (int kc = 0; kc < 4; ++kc) {
        int k0 = kc * 32 + half * 16;                // lane<16: K0..15 ; lane>=16: K16..31
        v8bf lo0 = *(const v8bf*)(ap0 + k0);
        v8bf hi0 = *(const v8bf*)(ap0 + k0 + 8);     // K+8
        v8bf lo1 = *(const v8bf*)(ap1 + k0);
        v8bf hi1 = *(const v8bf*)(ap1 + k0 + 8);
        afr0[kc] = __builtin_shufflevector(lo0, hi0, 0, 1, 2, 3, 4, 5, 6, 7,
                                                     8, 9, 10, 11, 12, 13, 14, 15);
        afr1[kc] = __builtin_shufflevector(lo1, hi1, 0, 1, 2, 3, 4, 5, 6, 7,
                                                     8, 9, 10, 11, 12, 13, 14, 15);
    }

#pragma unroll
    for (int nt = 0; nt < 8; ++nt) {
        const int colg = nt * 16 + l16;
        const float bz = bias[colg];
        v8f c0, c1;
#pragma unroll
        for (int r = 0; r < 8; ++r) { c0[r] = bz; c1[r] = bz; }   // C init = bias

#pragma unroll
        for (int kc = 0; kc < 4; ++kc) {
            // B fragment: lane (N=l16, K-half=half) reads 16 contiguous k values;
            // shared by both row tiles -> 2 independent WMMA chains per load.
            const unsigned short* bp = sB + colg * DH + kc * 32 + half * 16;
            v16bf bfr = *(const v16bf*)bp;           // 32 B, 32 B-aligned
            c0 = __builtin_amdgcn_wmma_f32_16x16x32_bf16(
                     false, afr0[kc], false, bfr, (short)0, c0, false, false);
            c1 = __builtin_amdgcn_wmma_f32_16x16x32_bf16(
                     false, afr1[kc], false, bfr, (short)0, c1, false, false);
        }

        // D layout: VGPR r -> M = half*8 + r, N = l16.
        float* zp0 = Z + (r0 + half * 8) * DH + colg;
        float* zp1 = Z + (r1 + half * 8) * DH + colg;
        if (fullblk) {                               // uniform: no exec masking
#pragma unroll
            for (int r = 0; r < 8; ++r) zp0[r * DH] = c0[r];
#pragma unroll
            for (int r = 0; r < 8; ++r) zp1[r * DH] = c1[r];
        } else {
#pragma unroll
            for (int r = 0; r < 8; ++r)
                if (r0 + half * 8 + r < N) zp0[r * DH] = c0[r];
#pragma unroll
            for (int r = 0; r < 8; ++r)
                if (r1 + half * 8 + r < N) zp1[r * DH] = c1[r];
        }
    }
}

// --------------------- per-column sum / sum-of-squares ---------------------
__global__ void k_stats(const float* __restrict__ Z, float* __restrict__ sums, int N) {
    const int c = threadIdx.x;                       // 128 threads = 128 columns
    long r0 = (long)blockIdx.x * 512;
    long r1 = r0 + 512; if (r1 > N) r1 = N;
    float s = 0.0f, s2 = 0.0f;
    for (long r = r0; r < r1; ++r) {
        float v = Z[r * DH + c];
        s += v; s2 += v * v;
    }
    atomicAdd(&sums[c], s);
    atomicAdd(&sums[DH + c], s2);
}

// ----------------------- BN + ReLU (-> bf16 / -> f32) ----------------------
__global__ void k_bn_relu_bf16(const float* __restrict__ Z, const float* __restrict__ sums,
                               const float* __restrict__ g, const float* __restrict__ be,
                               unsigned short* __restrict__ out, long n, float invN) {
    long i = (long)blockIdx.x * blockDim.x + threadIdx.x;
    if (i >= n) return;
    int c = (int)(i & 127);
    float mu  = sums[c] * invN;
    float var = sums[DH + c] * invN - mu * mu;
    float v = (Z[i] - mu) * rsqrtf(var + BN_EPS) * g[c] + be[c];
    out[i] = f2bf(v > 0.0f ? v : 0.0f);
}

__global__ void k_bn_relu_f32(const float* __restrict__ Z, const float* __restrict__ sums,
                              const float* __restrict__ g, const float* __restrict__ be,
                              float* __restrict__ out, long n, float invN) {
    long i = (long)blockIdx.x * blockDim.x + threadIdx.x;
    if (i >= n) return;
    int c = (int)(i & 127);
    float mu  = sums[c] * invN;
    float var = sums[DH + c] * invN - mu * mu;
    float v = (Z[i] - mu) * rsqrtf(var + BN_EPS) * g[c] + be[c];
    out[i] = v > 0.0f ? v : 0.0f;
}

// -------------------------- segment-sum pooling ----------------------------
__global__ void k_pool(const float* __restrict__ h, const int* __restrict__ gi,
                       float* __restrict__ pooled, int N) {
    long i = (long)blockIdx.x * blockDim.x + threadIdx.x;
    if (i >= (long)N * DH) return;
    int node = (int)(i >> 7), c = (int)(i & 127);
    atomicAdd(&pooled[(long)gi[node] * DH + c], h[i]);
}

// ---------------- logits += pooled @ W + b  (1024x128x10, tiny) ------------
__global__ void k_clf(const float* __restrict__ pooled, const float* __restrict__ W,
                      const float* __restrict__ b, float* __restrict__ logits, int NG) {
    int i = blockIdx.x * blockDim.x + threadIdx.x;
    if (i >= NG * 10) return;
    int g = i / 10, c = i % 10;
    float s = b[c];
    const float* p = pooled + (long)g * DH;
#pragma unroll 8
    for (int k = 0; k < DH; ++k) s += p[k] * W[k * 10 + c];
    logits[i] += s;
}

// ---------------------------------------------------------------------------
extern "C" void kernel_launch(void* const* d_in, const int* in_sizes, int n_in,
                              void* d_out, int out_size, void* d_ws, size_t ws_size,
                              hipStream_t stream) {
    const float* x  = (const float*)d_in[0];
    const int* row  = (const int*)d_in[1];
    const int* col  = (const int*)d_in[2];
    const int* gi   = (const int*)d_in[3];
    // d_in[4] = ng scalar (recomputed from out_size)

    const int N  = in_sizes[0] / DH;
    const int E  = in_sizes[1];
    const int NG = out_size / 10;
    const long NF = (long)N * DH;
    const float invN = 1.0f / (float)N;

    // layer l param j: {0:eps,1:W1,2:b1,3:g1,4:be1,5:W2,6:b2,7:g2,8:be2}
    auto LP = [&](int l, int j) { return (const float*)d_in[5 + l * 9 + j]; };
    // clf j param k: {0:W,1:b}
    auto CP = [&](int j, int k) { return (const float*)d_in[32 + j * 2 + k]; };

    char* ws = (char*)d_ws;
    float* h_buf = (float*)ws;                 ws += (size_t)NF * 4;
    float* zagg  = (float*)ws;                 ws += (size_t)NF * 4;   // agg, then GEMM out
    unsigned short* abuf = (unsigned short*)ws; ws += (size_t)NF * 2;  // bf16 GEMM A
    unsigned short* WT   = (unsigned short*)ws; ws += (size_t)DH * DH * 2;
    float* sums   = (float*)ws;                ws += 256 * 4;
    float* pooled = (float*)ws;                ws += (size_t)NG * DH * 4;

    float* logits = (float*)d_out;
    auto G = [](long n, int b) { return dim3((unsigned)((n + b - 1) / b)); };

    // logits = 0; pool layer-0 input (x) + classifier 0
    k_zero<<<G(out_size, 256), 256, 0, stream>>>(logits, out_size);
    k_zero<<<G((long)NG * DH, 256), 256, 0, stream>>>(pooled, (long)NG * DH);
    k_pool<<<G(NF, 256), 256, 0, stream>>>(x, gi, pooled, N);
    k_clf<<<G(NG * 10, 256), 256, 0, stream>>>(pooled, CP(0, 0), CP(0, 1), logits, NG);

    const float* h_cur = x;
    for (int l = 0; l < 3; ++l) {
        // agg = scatter-add of neighbor features
        k_zero<<<G(NF, 256), 256, 0, stream>>>(zagg, NF);
        k_scatter<<<G((long)E * 32, 256), 256, 0, stream>>>(h_cur, row, col, zagg, E);
        // z0 = (1+eps)*h + agg -> bf16
        k_fuse_eps<<<G(NF, 256), 256, 0, stream>>>(h_cur, zagg, LP(l, 0), abuf, NF);

        // GEMM1 (WMMA) + BN stats + BN/ReLU -> bf16
        k_cvt_wt<<<G(DH * DH, 256), 256, 0, stream>>>(LP(l, 1), WT);
        k_gemm128<<<dim3((N + 127) / 128), dim3(128), 0, stream>>>(abuf, WT, LP(l, 2), zagg, N);
        k_zero<<<1, 256, 0, stream>>>(sums, 256);
        k_stats<<<dim3((N + 511) / 512), dim3(128), 0, stream>>>(zagg, sums, N);
        k_bn_relu_bf16<<<G(NF, 256), 256, 0, stream>>>(zagg, sums, LP(l, 3), LP(l, 4),
                                                       abuf, NF, invN);

        // GEMM2 (WMMA) + BN stats + BN/ReLU -> f32 h
        k_cvt_wt<<<G(DH * DH, 256), 256, 0, stream>>>(LP(l, 5), WT);
        k_gemm128<<<dim3((N + 127) / 128), dim3(128), 0, stream>>>(abuf, WT, LP(l, 6), zagg, N);
        k_zero<<<1, 256, 0, stream>>>(sums, 256);
        k_stats<<<dim3((N + 511) / 512), dim3(128), 0, stream>>>(zagg, sums, N);
        k_bn_relu_f32<<<G(NF, 256), 256, 0, stream>>>(zagg, sums, LP(l, 7), LP(l, 8),
                                                      h_buf, NF, invN);
        h_cur = h_buf;

        // pooling + classifier for this layer's output
        k_zero<<<G((long)NG * DH, 256), 256, 0, stream>>>(pooled, (long)NG * DH);
        k_pool<<<G(NF, 256), 256, 0, stream>>>(h_cur, gi, pooled, N);
        k_clf<<<G(NG * 10, 256), 256, 0, stream>>>(pooled, CP(l + 1, 0), CP(l + 1, 1),
                                                   logits, NG);
    }
}
